// BiLSTM_CRF_14113262535036
// MI455X (gfx1250) — matrix operand order repaired and checked by reading, not compiled, not run on hardware
//
#include <hip/hip_runtime.h>
#include <hip/hip_bf16.h>

// ---------------------------------------------------------------------------
// BiLSTM-CRF for MI455X (gfx1250, wave32, WMMA).
// All heavy GEMMs use v_wmma_f32_16x16x32_bf16 with f32 accumulators.
// ---------------------------------------------------------------------------

typedef __attribute__((ext_vector_type(16))) __bf16 v16bf;
typedef __attribute__((ext_vector_type(8)))  float  v8f;

#define T_STEPS 128
#define BATCH   128
#define HID_H   256          // per-direction hidden
#define HID2    512          // bidirectional hidden
#define GATES   1024         // 4*H
#define KTAGS   60
#define START_TAG 58
#define STOP_TAG  59

__device__ __forceinline__ float sigm(float x) { return 1.f / (1.f + __expf(-x)); }
// fast tanh on v_exp_f32: tanh(x) = (e^{2x}-1)/(e^{2x}+1); |x|>~9 saturates in f32.
__device__ __forceinline__ float tanh_fast(float x) {
  float xc = fminf(fmaxf(x, -15.f), 15.f);
  float e = __expf(2.f * xc);
  return (e - 1.f) / (e + 1.f);
}

// ---------------------------------------------------------------------------
// fp32 -> bf16 conversion
// ---------------------------------------------------------------------------
__global__ __launch_bounds__(256) void cvt_bf16(const float* __restrict__ src,
                                                __bf16* __restrict__ dst, int n) {
  int i = blockIdx.x * 256 + threadIdx.x;
  if (i < n) dst[i] = (__bf16)src[i];
}

// relu(out) (+ prevRelu) -> curRelu (f32) and xb (bf16)
__global__ __launch_bounds__(256) void prep_x(const float* __restrict__ outp,
                                              const float* __restrict__ prevRelu,
                                              float* __restrict__ curRelu,
                                              __bf16* __restrict__ xb, int n) {
  int i = blockIdx.x * 256 + threadIdx.x;
  if (i < n) {
    float r = fmaxf(outp[i], 0.f);
    curRelu[i] = r;
    float s = r + (prevRelu ? prevRelu[i] : 0.f);
    xb[i] = (__bf16)s;
  }
}

// ---------------------------------------------------------------------------
// Layer-1 input projection: K = D_IN = 3, plain VALU (tiny).
// xw[dir][m][n] = b1[dir][n] + sum_d sent[m][d] * w1[dir][n][d]
// ---------------------------------------------------------------------------
__global__ __launch_bounds__(256) void proj1_kernel(const float* __restrict__ sent,
                                                    const float* __restrict__ w1,
                                                    const float* __restrict__ b1,
                                                    float* __restrict__ xw) {
  size_t idx = (size_t)blockIdx.x * 256 + threadIdx.x;   // < 2 * 16384 * 1024
  int dir = (int)(idx >> 24);
  int rem = (int)(idx & 0xFFFFFF);
  int m = rem >> 10, n = rem & 1023;
  const float* wr = w1 + ((size_t)dir * GATES + n) * 3;
  const float* xr = sent + (size_t)m * 3;
  xw[idx] = b1[dir * GATES + n] + xr[0] * wr[0] + xr[1] * wr[1] + xr[2] * wr[2];
}

// ---------------------------------------------------------------------------
// Generic WMMA GEMM:  Y[M,N] = X[M,K] * W[N,K]^T + bias[N]
//   X, W bf16 row-major; Y f32 row-major. M % 128 == 0, K % 32 == 0, any N.
//   Block: 256 threads (8 waves); block tile 128x128; wave tile 32x64.
//   LDS tiles stored in WMMA fragment order:
//     A frag (16x32): lane l -> row (l&15); elem e -> k = (l>>4)*16 + e
//     B frag (32x16): lane l -> col (l&15); elem e -> k = (l>>4)*16 + e
// ---------------------------------------------------------------------------
__global__ __launch_bounds__(256) void gemm_bf16(const __bf16* __restrict__ X,
                                                 const __bf16* __restrict__ W,
                                                 const float* __restrict__ bias,
                                                 float* __restrict__ Y,
                                                 int M, int N, int K) {
  __shared__ __align__(32) __bf16 ldsA[8 * 32 * 16];   // 8 KB
  __shared__ __align__(32) __bf16 ldsB[8 * 32 * 16];   // 8 KB

  const int tid  = threadIdx.x;
  const int lane = tid & 31;
  const int w    = tid >> 5;
  const int half = lane >> 4;
  const int ncol = lane & 15;
  const int m0 = blockIdx.x * 128;
  const int n0 = blockIdx.y * 128;
  const int mts = (w & 3) * 2;        // 2 m-tiles per wave
  const int nts = (w >> 2) * 4;       // 4 n-tiles per wave

  // accumulators seeded with bias
  v8f acc[2][4];
#pragma unroll
  for (int nt = 0; nt < 4; ++nt) {
    int col = n0 + (nts + nt) * 16 + ncol;
    float bv = (col < N) ? bias[col] : 0.f;
#pragma unroll
    for (int mt = 0; mt < 2; ++mt)
#pragma unroll
      for (int v = 0; v < 8; ++v) acc[mt][nt][v] = bv;
  }

  // cooperative load role: tid -> (tile lt 0..7, lane)
  const int lt = w;
  const __bf16* xrow = X + (size_t)(m0 + lt * 16 + ncol) * K;
  const int nB = n0 + lt * 16 + ncol;
  const __bf16* wrow = (nB < N) ? (W + (size_t)nB * K) : nullptr;
  unsigned int* dA = (unsigned int*)(ldsA + ((lt * 32 + lane) << 4));
  __bf16* dB = ldsB + ((lt * 32 + lane) << 4);

  for (int kk = 0; kk < K; kk += 32) {
    __syncthreads();
    // A tile in fragment order (2 bf16 per u32, pairs contiguous in memory)
#pragma unroll
    for (int v = 0; v < 8; ++v) {
      int k = (half << 4) + (v << 1);
      dA[v] = *(const unsigned int*)(xrow + kk + k);
    }
    // B tile: 16 contiguous K values per lane
    if (wrow) {
      *(v16bf*)dB = *(const v16bf*)(wrow + kk + (half << 4));
    } else {
#pragma unroll
      for (int e = 0; e < 16; ++e) dB[e] = (__bf16)0.f;
    }
    __syncthreads();

    v16bf a[2];
#pragma unroll
    for (int mt = 0; mt < 2; ++mt)
      a[mt] = *(const v16bf*)(ldsA + (((mts + mt) * 32 + lane) << 4));
#pragma unroll
    for (int nt = 0; nt < 4; ++nt) {
      v16bf bmat = *(const v16bf*)(ldsB + (((nts + nt) * 32 + lane) << 4));
#pragma unroll
      for (int mt = 0; mt < 2; ++mt)
        acc[mt][nt] = __builtin_amdgcn_wmma_f32_16x16x32_bf16(
            false, a[mt], false, bmat, (short)0, acc[mt][nt], false, false);
    }
  }

  // store (D layout: row = v + 8*half, col = lane&15)
#pragma unroll
  for (int mt = 0; mt < 2; ++mt)
#pragma unroll
    for (int nt = 0; nt < 4; ++nt) {
      int col = n0 + (nts + nt) * 16 + ncol;
      if (col < N) {
        int rb = m0 + (mts + mt) * 16 + half * 8;
#pragma unroll
        for (int v = 0; v < 8; ++v)
          Y[(size_t)(rb + v) * N + col] = acc[mt][nt][v];
      }
    }
}

// ---------------------------------------------------------------------------
// LSTM recurrence (one workgroup per direction, grid.x = 2).
// 512 threads = 16 waves; wave w owns j-tile w (16 of the 256 hidden cols).
// Per step: gates[128,1024] = xw_t + h[128,256] @ Whh^T, then cell update.
// h kept in LDS in A-fragment order, double-buffered (2 x 64KB);
// c kept in LDS as f32 (128KB). Total dynamic LDS = 256KB (<320KB/WGP).
// Whh (512KB bf16, L2-resident, reused 128x) streamed as B fragments.
// ---------------------------------------------------------------------------
__global__ __launch_bounds__(512) void rec_lstm(const float* __restrict__ xwBase,
                                                const __bf16* __restrict__ whhBase,
                                                const float* __restrict__ h0L,
                                                const float* __restrict__ c0L,
                                                float* __restrict__ outBase,
                                                int T) {
  const int dir = blockIdx.x;
  const float*  xw  = xwBase + (size_t)dir * T * BATCH * GATES;
  const __bf16* whh = whhBase + (size_t)dir * GATES * HID_H;
  const float*  h0  = h0L + (size_t)dir * BATCH * HID_H;
  const float*  c0  = c0L + (size_t)dir * BATCH * HID_H;
  const int outOff  = dir * HID_H;

  extern __shared__ char smem_raw[];
  char* smem = (char*)(((size_t)smem_raw + 31) & ~(size_t)31);
  __bf16* hbuf0 = (__bf16*)smem;                 // 8mt x 8kt x 32lane x 16elem
  __bf16* hbuf1 = hbuf0 + 8 * 8 * 32 * 16;
  float*  cLds  = (float*)(smem + 2 * 8 * 8 * 32 * 16 * sizeof(__bf16));

  const int tid  = threadIdx.x;
  const int w    = tid >> 5;       // 0..15 == j-tile
  const int lane = tid & 31;
  const int half = lane >> 4;
  const int ncol = lane & 15;
  const int jt   = w;
  const int j    = jt * 16 + ncol;

  // init c
  for (int i = tid; i < BATCH * HID_H; i += 512) cLds[i] = c0[i];
  // init h fragment buffer 0 from h0 (f32 -> bf16, fragment-order swizzle)
  for (int s = tid; s < 2048; s += 512) {
    int mt = s >> 8, kt = (s >> 5) & 7, l = s & 31;
    int b = mt * 16 + (l & 15), hf = l >> 4;
    __bf16* dst = hbuf0 + s * 16;
    const float* srcrow = h0 + (size_t)b * HID_H + kt * 32;
#pragma unroll
    for (int e = 0; e < 16; ++e) {
      int k = (hf << 4) + e;
      dst[e] = (__bf16)srcrow[k];
    }
  }
  __syncthreads();

  for (int t = 0; t < T; ++t) {
    const int tt = dir ? (T - 1 - t) : t;
    const __bf16* hcur = (t & 1) ? hbuf1 : hbuf0;
    __bf16*       hnxt = (t & 1) ? hbuf0 : hbuf1;
    const float* xwt  = xw + (size_t)tt * BATCH * GATES;
    float*       outt = outBase + (size_t)tt * BATCH * HID2 + outOff;

    for (int g2 = 0; g2 < 2; ++g2) {          // two groups of 4 m-tiles
      const int mts = g2 * 4;
      v8f acc[4][4];                           // [m-tile][gate]
      // seed accumulators with precomputed input projection (C operand)
#pragma unroll
      for (int mt = 0; mt < 4; ++mt) {
        int brow = (mts + mt) * 16 + half * 8;
#pragma unroll
        for (int g = 0; g < 4; ++g) {
          int col = g * HID_H + j;
#pragma unroll
          for (int v = 0; v < 8; ++v)
            acc[mt][g][v] = xwt[(size_t)(brow + v) * GATES + col];
        }
      }
      // K = 256 -> 8 k-steps of 32
      for (int kt = 0; kt < 8; ++kt) {
        v16bf a[4];
#pragma unroll
        for (int mt = 0; mt < 4; ++mt)
          a[mt] = *(const v16bf*)(hcur + ((((mts + mt) * 8 + kt) * 32 + lane) << 4));
#pragma unroll
        for (int g = 0; g < 4; ++g) {
          int n = g * HID_H + j;                      // Whh row
          v16bf bmat = *(const v16bf*)(whh + (size_t)n * HID_H + kt * 32 + (half << 4));
#pragma unroll
          for (int mt = 0; mt < 4; ++mt)
            acc[mt][g] = __builtin_amdgcn_wmma_f32_16x16x32_bf16(
                false, a[mt], false, bmat, (short)0, acc[mt][g], false, false);
        }
      }
      // cell update (gate order i,f,g,o); D layout: row = v + 8*half
#pragma unroll
      for (int mt = 0; mt < 4; ++mt) {
        int b0 = (mts + mt) * 16 + half * 8;
#pragma unroll
        for (int v = 0; v < 8; ++v) {
          int b = b0 + v;
          float ig = sigm(acc[mt][0][v]);
          float fg = sigm(acc[mt][1][v]);
          float gg = tanh_fast(acc[mt][2][v]);
          float og = sigm(acc[mt][3][v]);
          float c = fg * cLds[b * HID_H + j] + ig * gg;
          cLds[b * HID_H + j] = c;
          float h = og * tanh_fast(c);
          outt[(size_t)b * HID2 + j] = h;
          // scatter h (bf16) into next A-fragment buffer
          int kt2 = j >> 5, kin = j & 31;
          int h2 = kin >> 4;
          int e2 = kin & 15;
          int lane2 = (b & 15) + (h2 << 4);
          hnxt[((((b >> 4) * 8 + kt2) * 32 + lane2) << 4) + e2] = (__bf16)h;
        }
      }
    }
    __syncthreads();   // hnxt complete before it becomes hcur
  }
}

// ---------------------------------------------------------------------------
// CRF negative log-likelihood: one workgroup per batch element.
// ---------------------------------------------------------------------------
__global__ __launch_bounds__(64) void crf_nll(const float* __restrict__ feats,
                                              const int* __restrict__ tags,
                                              const float* __restrict__ trans,
                                              float* __restrict__ out, int T) {
  const int b = blockIdx.x, tid = threadIdx.x;
  __shared__ float sTr[KTAGS * KTAGS];
  __shared__ float sAl[KTAGS];
  for (int i = tid; i < KTAGS * KTAGS; i += 64) sTr[i] = trans[i];
  if (tid < KTAGS) sAl[tid] = (tid == START_TAG) ? 0.f : -10000.f;
  __syncthreads();

  for (int t = 0; t < T; ++t) {
    float nv = 0.f;
    if (tid < KTAGS) {
      float feat = feats[((size_t)t * BATCH + b) * KTAGS + tid];
      const float* tr = sTr + tid * KTAGS;
      float m = -3.4e38f;
      for (int p = 0; p < KTAGS; ++p) m = fmaxf(m, sAl[p] + tr[p]);
      float s = 0.f;
      for (int p = 0; p < KTAGS; ++p) s += __expf(sAl[p] + tr[p] - m);
      nv = m + __logf(s) + feat;
    }
    __syncthreads();
    if (tid < KTAGS) sAl[tid] = nv;
    __syncthreads();
  }

  if (tid == 0) {
    const float* trS = sTr + STOP_TAG * KTAGS;
    float m = -3.4e38f;
    for (int p = 0; p < KTAGS; ++p) m = fmaxf(m, sAl[p] + trS[p]);
    float s = 0.f;
    for (int p = 0; p < KTAGS; ++p) s += __expf(sAl[p] + trS[p] - m);
    float logz = m + __logf(s);

    float gold = 0.f;
    int prev = START_TAG;
    for (int t = 0; t < T; ++t) {
      int cur = tags[t * BATCH + b];
      gold += sTr[cur * KTAGS + prev] + feats[((size_t)t * BATCH + b) * KTAGS + cur];
      prev = cur;
    }
    gold += sTr[STOP_TAG * KTAGS + prev];
    out[b] = logz - gold;
  }
}

// ---------------------------------------------------------------------------
// Host launcher
// ---------------------------------------------------------------------------
extern "C" void kernel_launch(void* const* d_in, const int* in_sizes, int n_in,
                              void* d_out, int out_size, void* d_ws, size_t ws_size,
                              hipStream_t stream) {
  (void)in_sizes; (void)n_in; (void)out_size; (void)ws_size;
  const float* sent   = (const float*)d_in[0];
  const int*   tags   = (const int*)d_in[1];
  const float* w_ih1  = (const float*)d_in[2];
  const float* w_hh1  = (const float*)d_in[3];
  const float* b1     = (const float*)d_in[4];
  const float* w_ih   = (const float*)d_in[5];
  const float* w_hh   = (const float*)d_in[6];
  const float* bias_l = (const float*)d_in[7];
  const float* fc_w   = (const float*)d_in[8];
  const float* fc_b   = (const float*)d_in[9];
  const float* h0     = (const float*)d_in[10];
  const float* c0     = (const float*)d_in[11];
  const float* trans  = (const float*)d_in[12];
  float* out = (float*)d_out;

  char* ws = (char*)d_ws;
  size_t off = 0;
  auto alloc = [&](size_t bytes) -> char* {
    char* p = ws + off;
    off = (off + bytes + 255) & ~(size_t)255;
    return p;
  };

  const size_t M = (size_t)T_STEPS * BATCH;  // 16384
  __bf16* whh1_bf = (__bf16*)alloc(2 * GATES * HID_H * 2);
  __bf16* wih_bf  = (__bf16*)alloc((size_t)6 * 2 * GATES * HID2 * 2);
  __bf16* whh_bf  = (__bf16*)alloc((size_t)6 * 2 * GATES * HID_H * 2);
  __bf16* fcw_bf  = (__bf16*)alloc((size_t)KTAGS * HID2 * 2);
  __bf16* xb      = (__bf16*)alloc(M * HID2 * 2);
  float*  xw      = (float*)alloc(2 * M * GATES * 4);
  float*  outBuf  = (float*)alloc(M * HID2 * 4);
  float*  reluA   = (float*)alloc(M * HID2 * 4);
  float*  reluB   = (float*)alloc(M * HID2 * 4);
  float*  feats   = (float*)alloc(M * KTAGS * 4);

  const int recLds = 2 * 8 * 8 * 32 * 16 * 2 /*h bufs*/ + BATCH * HID_H * 4 /*c*/ + 32;

  // weight conversion
  { int n = 2 * GATES * HID_H;        cvt_bf16<<<(n + 255) / 256, 256, 0, stream>>>(w_hh1, whh1_bf, n); }
  { int n = 6 * 2 * GATES * HID2;     cvt_bf16<<<(n + 255) / 256, 256, 0, stream>>>(w_ih,  wih_bf,  n); }
  { int n = 6 * 2 * GATES * HID_H;    cvt_bf16<<<(n + 255) / 256, 256, 0, stream>>>(w_hh,  whh_bf,  n); }
  { int n = KTAGS * HID2;             cvt_bf16<<<(n + 255) / 256, 256, 0, stream>>>(fc_w,  fcw_bf,  n); }

  // layer 1: tiny-K projection + recurrence
  proj1_kernel<<<(2u * M * GATES) / 256, 256, 0, stream>>>(sent, w_ih1, b1, xw);
  rec_lstm<<<2, 512, recLds, stream>>>(xw, whh1_bf, h0, c0, outBuf, T_STEPS);

  const int nElem = (int)(M * HID2);
  prep_x<<<(nElem + 255) / 256, 256, 0, stream>>>(outBuf, nullptr, reluA, xb, nElem);

  float* curR = reluA;
  for (int l = 0; l < 6; ++l) {
    for (int dir = 0; dir < 2; ++dir) {
      gemm_bf16<<<dim3((unsigned)(M / 128), GATES / 128), 256, 0, stream>>>(
          xb, wih_bf + (size_t)(l * 2 + dir) * GATES * HID2,
          bias_l + (size_t)(l * 2 + dir) * GATES,
          xw + (size_t)dir * M * GATES, (int)M, GATES, HID2);
    }
    rec_lstm<<<2, 512, recLds, stream>>>(
        xw, whh_bf + (size_t)l * 2 * GATES * HID_H,
        h0 + (size_t)(l + 1) * 2 * BATCH * HID_H,
        c0 + (size_t)(l + 1) * 2 * BATCH * HID_H, outBuf, T_STEPS);
    if (l < 5) {
      float* nxt = (curR == reluA) ? reluB : reluA;
      prep_x<<<(nElem + 255) / 256, 256, 0, stream>>>(outBuf, curR, nxt, xb, nElem);
      curR = nxt;
    }
  }

  // FC: feats = out7 @ fc_w^T + fc_b   (N = 60, guarded)
  cvt_bf16<<<(nElem + 255) / 256, 256, 0, stream>>>(outBuf, xb, nElem);
  gemm_bf16<<<dim3((unsigned)(M / 128), 1), 256, 0, stream>>>(
      xb, fcw_bf, fc_b, feats, (int)M, KTAGS, HID2);

  // CRF NLL
  crf_nll<<<BATCH, 64, 0, stream>>>(feats, tags, trans, out, T_STEPS);
}
